// TGCSRN_27444841021716
// MI455X (gfx1250) — matrix operand order repaired and checked by reading, not compiled, not run on hardware
//
#include <hip/hip_runtime.h>
#include <hip/hip_bf16.h>
#include <math.h>

typedef _Float16 hv16 __attribute__((ext_vector_type(16)));
typedef _Float16 hv8  __attribute__((ext_vector_type(8)));
typedef float    fv8  __attribute__((ext_vector_type(8)));

#define NNODE 500
#define NBATCH 16
#define NEXP 10
#define MAXTILES 48
#define SLOTH 16384  // halves per packed weight slot

// ---------------- WMMA helpers ----------------
__device__ inline fv8 wmma_f16(hv16 a, hv16 b, fv8 c) {
  return __builtin_amdgcn_wmma_f32_16x16x32_f16(false, a, false, b, (short)0, c, false, false);
}

// A fragment (16xK tile, row-major f16, bounds-checked). contiguous k order (same as load_b_gen):
// lanes 0-15: m=lane, halves 0-15 -> k0..k0+15
// lanes 16-31: m=lane-16, halves 0-15 -> k0+16..k0+31
__device__ inline hv16 load_a_gen(const _Float16* A, int lda, int m0, int k0,
                                  int M, int K, int lane) {
  int m = m0 + (lane & 15);
  bool mv = m < M;
  const _Float16* row = A + (size_t)m * lda;
  int kb1 = k0 + ((lane & 16) ? 16 : 0);
  int kb2 = kb1 + 8;
  hv16 a;
#pragma unroll
  for (int i = 0; i < 8; ++i) { int k = kb1 + i; a[i]   = (mv && k < K) ? row[k] : (_Float16)0.f; }
#pragma unroll
  for (int i = 0; i < 8; ++i) { int k = kb2 + i; a[8+i] = (mv && k < K) ? row[k] : (_Float16)0.f; }
  return a;
}

// B fragment (KxN row-major f16, bounds-checked). contiguous k order (same as load_a_gen):
// lanes 0-15: n=lane, halves i -> k0+i ; lanes 16-31: n=lane-16, halves i -> k0+16+i
__device__ inline hv16 load_b_gen(const _Float16* B, int ldb, int k0, int n0,
                                  int K, int N, int lane) {
  int n = n0 + (lane & 15);
  bool nv = n < N;
  int kb = k0 + ((lane & 16) ? 16 : 0);
  hv16 b;
#pragma unroll
  for (int i = 0; i < 16; ++i) { int k = kb + i; b[i] = (nv && k < K) ? B[(size_t)k * ldb + n] : (_Float16)0.f; }
  return b;
}

// A fragment from LDS, 16x128 row-major, vectorized (16B loads)
__device__ inline hv16 lds_a(const _Float16* base, int kt, int lane) {
  const _Float16* row = base + (lane & 15) * 128;
  int k0 = kt * 32;
  int kb1 = k0 + ((lane & 16) ? 16 : 0);
  int kb2 = kb1 + 8;
  hv8 lo = *(const hv8*)(row + kb1);
  hv8 hi = *(const hv8*)(row + kb2);
  return __builtin_shufflevector(lo, hi, 0,1,2,3,4,5,6,7,8,9,10,11,12,13,14,15);
}

// B fragment from packed weights: lane reads 32 contiguous bytes
__device__ inline hv16 ld_bpk(const _Float16* slot, int kt, int nt, int NT, int lane) {
  const _Float16* p = slot + (((size_t)(kt * NT + nt) * 32 + lane) << 4);
  hv8 lo = *(const hv8*)p;
  hv8 hi = *(const hv8*)(p + 8);
  return __builtin_shufflevector(lo, hi, 0,1,2,3,4,5,6,7,8,9,10,11,12,13,14,15);
}

// ---------------- small kernels ----------------
// mat_k = env_cb @ w_k (10x64), mat_v = env_cb @ w_v (10x10)
__global__ void k_small0(const float* env, const float* wk, const float* wv,
                         float* matk, float* matv) {
  int tid = threadIdx.x;
  if (tid < 640) {
    int j = tid >> 6, c = tid & 63;
    float s = 0.f;
    for (int d = 0; d < 64; ++d) s += env[j*64+d] * wk[d*64+c];
    matk[tid] = s;
  } else if (tid < 740) {
    int t2 = tid - 640, j = t2 / 10, c = t2 % 10;
    float s = 0.f;
    for (int d = 0; d < 64; ++d) s += env[j*64+d] * wv[d*10+c];
    matv[j*10+c] = s;
  }
}

// per-node expert selection (argmax of logits; fwd value of mask is one-hot)
__global__ void k_sel(const float* ncb, const float* wq, const float* matk,
                      const float* matv, int* sel) {
  int n = blockIdx.x * 256 + threadIdx.x;
  if (n >= NNODE) return;
  float q[64];
  for (int c = 0; c < 64; ++c) {
    float s = 0.f;
    for (int d = 0; d < 64; ++d) s += ncb[n*64+d] * wq[d*64+c];
    q[c] = s;
  }
  float sc[10]; float mx = -1e30f;
  for (int j = 0; j < 10; ++j) {
    float s = 0.f;
    for (int c = 0; c < 64; ++c) s += q[c] * matk[j*64+c];
    s *= 0.125f;
    sc[j] = s; if (s > mx) mx = s;
  }
  float sum = 0.f;
  for (int j = 0; j < 10; ++j) { sc[j] = expf(sc[j]-mx); sum += sc[j]; }
  float inv = 1.f / sum;
  float best = -1e30f; int bi = 0;
  for (int j = 0; j < 10; ++j) {
    float lg = 0.f;
    for (int i = 0; i < 10; ++i) lg += sc[i]*inv * matv[i*10+j];
    if (lg > best) { best = lg; bi = j; }
  }
  sel[n] = bi;
}

// build per-expert 16-node tiles
__global__ void k_tiles(const int* sel, int* tile_expert, int* tile_nodes, int* ntiles) {
  __shared__ int cnt[NEXP], tbase[NEXP], fill[NEXP];
  int tid = threadIdx.x;
  if (tid < NEXP) { cnt[tid] = 0; fill[tid] = 0; }
  __syncthreads();
  if (tid < NNODE) atomicAdd(&cnt[sel[tid]], 1);
  __syncthreads();
  if (tid == 0) {
    int t = 0;
    for (int e = 0; e < NEXP; ++e) {
      tbase[e] = t;
      int nt = (cnt[e] + 15) >> 4;
      for (int k = 0; k < nt; ++k) if (t + k < MAXTILES) tile_expert[t+k] = e;
      t += nt;
    }
    *ntiles = t;
  }
  __syncthreads();
  for (int i = tid; i < MAXTILES*16; i += 512) tile_nodes[i] = -1;
  __syncthreads();
  if (tid < NNODE) {
    int e = sel[tid];
    int p = atomicAdd(&fill[e], 1);
    tile_nodes[tbase[e]*16 + p] = tid;
  }
}

// timestep embedding (B,T,64), also f16 copy of t==11 row
__global__ void k_temb(const int* tpos, float* temb, _Float16* temb11h) {
  int idx = blockIdx.x * 256 + threadIdx.x;
  if (idx >= NBATCH*12*64) return;
  int j = idx & 63, t = (idx >> 6) % 12, b = idx / (64*12);
  int p0 = tpos[(b*12+t)*2], p1 = tpos[(b*12+t)*2+1];
  float tp = (p0 <= 4) ? (float)p1 : (float)(23 + p1);
  int jj = j & 31;
  float fr = expf(-logf(10000.f) * (float)jj / 32.f);
  float a = tp * fr;
  float v = (j < 32) ? cosf(a) : sinf(a);
  temb[idx] = v;
  if (t == 11) temb11h[b*64+j] = (_Float16)v;
}

// total_x, timesteps 0..4 only, f16: layout [b][n][t][128]
__global__ void k_totalx(const float* x, const float* fcxW, const float* fcxb,
                         const float* temb, _Float16* X5) {
  long long idx = (long long)blockIdx.x * 256 + threadIdx.x;
  if (idx >= (long long)NBATCH*NNODE*5*128) return;
  int c = idx & 127;
  long long r = idx >> 7;
  int t = r % 5;
  long long bn = r / 5;
  int n = bn % NNODE, b = bn / NNODE;
  float v;
  if (c < 64) {
    const float* xe = x + (((size_t)(b*NNODE+n)*12 + t) << 1);
    v = xe[0]*fcxW[c] + xe[1]*fcxW[64+c] + fcxb[c];
  } else {
    v = temb[(b*12+t)*64 + (c-64)];
  }
  X5[idx] = (_Float16)v;
}

// pack TCN weight matrices into per-lane fragment layout (f16)
__global__ void k_pack(const float* WinN, const float* WfN, const float* WgN, const float* WoN,
                       const float* WinP, const float* WfP, const float* WgP, const float* WoP,
                       _Float16* PW) {
  int slot = blockIdx.x, tcn = blockIdx.y, e = blockIdx.z;
  const float* Win = tcn ? WinP : WinN;
  const float* Wf  = tcn ? WfP  : WfN;
  const float* Wg  = tcn ? WgP  : WgN;
  const float* Wo  = tcn ? WoP  : WoN;
  const float* src; int O;
  if (slot == 0)      { src = Win + (size_t)e*16384; O = 128; }
  else if (slot <= 4) { src = Wf + (size_t)(e*4 + (slot-1))*16384; O = 128; }
  else if (slot <= 8) { src = Wg + (size_t)(e*4 + (slot-5))*16384; O = 128; }
  else                { src = Wo + (size_t)e*8192; O = 64; }
  int NT = O >> 4;
  unsigned int* dst = (unsigned int*)(PW + (size_t)((e*2+tcn)*10 + slot) * SLOTH);
  int tot = 1024 * NT;
  for (int d = threadIdx.x; d < tot; d += 256) {
    int w = d & 7, lane = (d >> 3) & 31, f = d >> 8;
    int nt = f % NT, kt = f / NT;
    int n = nt*16 + (lane & 15);
    int kb = kt*32 + ((lane & 16) ? 16 : 0) + 2*w;
    union { _Float16 h[2]; unsigned int u; } pk;
    pk.h[0] = (_Float16)src[(size_t)kb*O + n];
    pk.h[1] = (_Float16)src[(size_t)(kb+1)*O + n];
    dst[d] = pk.u;
  }
}

// ---------------- TCN (the big WMMA kernel) ----------------
__global__ __launch_bounds__(128) void k_tcn(
    const _Float16* __restrict__ X5, const _Float16* __restrict__ PW,
    const int* __restrict__ tile_expert, const int* __restrict__ tile_nodes,
    const int* __restrict__ ntiles,
    const float* binN, const float* bfN, const float* bgN, const float* boutN,
    const float* binP, const float* bfP, const float* bgP, const float* boutP,
    float* __restrict__ unb, float* __restrict__ hpv) {
  if ((int)blockIdx.x >= ntiles[0]) return;
  int b = blockIdx.y;
  int e = tile_expert[blockIdx.x];
  int tid = threadIdx.x, lane = tid & 31, wave = tid >> 5;
  __shared__ int snode[16];
  __shared__ __align__(16) _Float16 sX[5][16][128];
  __shared__ __align__(16) _Float16 sH[4][16][128];
  __shared__ __align__(16) _Float16 sHn[2][16][128];
  __shared__ __align__(16) _Float16 sH2[16][128];
  if (tid < 16) snode[tid] = tile_nodes[blockIdx.x*16 + tid];
  __syncthreads();
  // stage X tile (5 timesteps x 16 rows x 128) into LDS
  const unsigned int* src32 = (const unsigned int*)X5;
  for (int i = tid; i < 5*16*64; i += 128) {
    int cp = i & 63, m = (i >> 6) & 15, t = i >> 10;
    int node = snode[m];
    unsigned int v = 0u;
    if (node >= 0) v = src32[(((size_t)(b*NNODE+node)*5 + t) << 6) + cp];
    ((unsigned int*)&sX[t][m][0])[cp] = v;
  }
  __syncthreads();

  int col = lane & 15, rb = (lane & 16) ? 8 : 0;
  for (int phase = 0; phase < 2; ++phase) {
    int toff = (phase == 0) ? 1 : 0;
    const _Float16* slotBase = PW + (size_t)((e*2 + phase) * 10) * SLOTH;
    const float* bin  = phase ? binP  : binN;
    const float* bf   = phase ? bfP   : bfN;
    const float* bg   = phase ? bgP   : bgN;
    const float* bout = phase ? boutP : boutN;

    // step1: h0[t] = X[t+toff] @ Win + bin, t=0..3
    for (int t = 0; t < 4; ++t) {
      const _Float16* A = &sX[t + toff][0][0];
      for (int sub = 0; sub < 2; ++sub) {
        int nt = wave + 4*sub; int cg = nt*16 + col;
        float bv = bin[e*128 + cg];
        fv8 acc;
#pragma unroll
        for (int r = 0; r < 8; ++r) acc[r] = bv;
#pragma unroll
        for (int kt = 0; kt < 4; ++kt)
          acc = wmma_f16(lds_a(A, kt, lane), ld_bpk(slotBase, kt, nt, 8, lane), acc);
#pragma unroll
        for (int r = 0; r < 8; ++r) sH[t][rb + r][cg] = (_Float16)acc[r];
      }
    }
    __syncthreads();

    // step2: block0 (d=1), positions 0 and 2
    for (int pi = 0; pi < 2; ++pi) {
      int tp = pi * 2;
      for (int sub = 0; sub < 2; ++sub) {
        int nt = wave + 4*sub; int cg = nt*16 + col;
        float bfv = bf[(e*2 + 0)*128 + cg];
        float bgv = bg[(e*2 + 0)*128 + cg];
        fv8 af, ag;
#pragma unroll
        for (int r = 0; r < 8; ++r) { af[r] = bfv; ag[r] = bgv; }
        for (int j = 0; j < 2; ++j) {
          const _Float16* A  = &sH[tp + j][0][0];
          const _Float16* Wf = slotBase + (size_t)(1 + j) * SLOTH;
          const _Float16* Wg = slotBase + (size_t)(5 + j) * SLOTH;
#pragma unroll
          for (int kt = 0; kt < 4; ++kt) {
            hv16 a = lds_a(A, kt, lane);
            af = wmma_f16(a, ld_bpk(Wf, kt, nt, 8, lane), af);
            ag = wmma_f16(a, ld_bpk(Wg, kt, nt, 8, lane), ag);
          }
        }
#pragma unroll
        for (int r = 0; r < 8; ++r) {
          float f = tanhf(af[r]);
          float g = 1.f / (1.f + expf(-ag[r]));
          float pv = (float)sH[tp + 1][rb + r][cg];
          sHn[pi][rb + r][cg] = (_Float16)(f * g + pv);
        }
      }
    }
    __syncthreads();

    // step3: block1 (d=2), position 0 using hn[0], hn[2]
    for (int sub = 0; sub < 2; ++sub) {
      int nt = wave + 4*sub; int cg = nt*16 + col;
      float bfv = bf[(e*2 + 1)*128 + cg];
      float bgv = bg[(e*2 + 1)*128 + cg];
      fv8 af, ag;
#pragma unroll
      for (int r = 0; r < 8; ++r) { af[r] = bfv; ag[r] = bgv; }
      for (int j = 0; j < 2; ++j) {
        const _Float16* A  = &sHn[j][0][0];
        const _Float16* Wf = slotBase + (size_t)(3 + j) * SLOTH;
        const _Float16* Wg = slotBase + (size_t)(7 + j) * SLOTH;
#pragma unroll
        for (int kt = 0; kt < 4; ++kt) {
          hv16 a = lds_a(A, kt, lane);
          af = wmma_f16(a, ld_bpk(Wf, kt, nt, 8, lane), af);
          ag = wmma_f16(a, ld_bpk(Wg, kt, nt, 8, lane), ag);
        }
      }
#pragma unroll
      for (int r = 0; r < 8; ++r) {
        float f = tanhf(af[r]);
        float g = 1.f / (1.f + expf(-ag[r]));
        float pv = (float)sHn[1][rb + r][cg];
        sH2[rb + r][cg] = (_Float16)(f * g + pv);
      }
    }
    __syncthreads();

    // step4: out = h2 @ Wout + bout  (O = 64 -> 4 n-tiles, one per wave)
    {
      int nt = wave; int cg = nt*16 + col;
      float bv = bout[e*64 + cg];
      fv8 acc;
#pragma unroll
      for (int r = 0; r < 8; ++r) acc[r] = bv;
      const _Float16* Wo = slotBase + (size_t)9 * SLOTH;
#pragma unroll
      for (int kt = 0; kt < 4; ++kt)
        acc = wmma_f16(lds_a(&sH2[0][0], kt, lane), ld_bpk(Wo, kt, nt, 4, lane), acc);
      float* dst = phase ? hpv : unb;
#pragma unroll
      for (int r = 0; r < 8; ++r) {
        int node = snode[rb + r];
        if (node >= 0) dst[((size_t)(b*NNODE + node)) * 64 + cg] = acc[r];
      }
    }
    __syncthreads();
  }
}

// ---------------- generic 1-wave WMMA GEMM ----------------
__global__ __launch_bounds__(32) void k_gemm(
    const _Float16* __restrict__ A, long long sA, int lda,
    const _Float16* __restrict__ B, long long sB, int ldb,
    float* __restrict__ C, long long sC, int ldc,
    const float* __restrict__ bias, long long sBias,
    int M, int N, int K, int beta) {
  int lane = threadIdx.x;
  int m0 = blockIdx.x * 16, n0 = blockIdx.y * 16;
  int bt = blockIdx.z;
  A += (size_t)bt * sA; B += (size_t)bt * sB; C += (size_t)bt * sC;
  int col = lane & 15, rb = (lane & 16) ? 8 : 0;
  fv8 acc;
  if (beta) {
#pragma unroll
    for (int r = 0; r < 8; ++r) {
      int m = m0 + rb + r, n = n0 + col;
      acc[r] = (m < M && n < N) ? C[(size_t)m * ldc + n] : 0.f;
    }
  } else {
#pragma unroll
    for (int r = 0; r < 8; ++r) acc[r] = 0.f;
  }
  int KT = (K + 31) >> 5;
  for (int kt = 0; kt < KT; ++kt) {
    hv16 a = load_a_gen(A, lda, m0, kt * 32, M, K, lane);
    hv16 bfr = load_b_gen(B, ldb, kt * 32, n0, K, N, lane);
    acc = wmma_f16(a, bfr, acc);
  }
  float bv = 0.f;
  if (bias) { int n = n0 + col; if (n < N) bv = bias[(size_t)bt * sBias + n]; }
#pragma unroll
  for (int r = 0; r < 8; ++r) {
    int m = m0 + rb + r, n = n0 + col;
    if (m < M && n < N) C[(size_t)m * ldc + n] = acc[r] + bv;
  }
}

// ---------------- misc elementwise / reductions ----------------
__global__ void k_f32h(_Float16* dst, const float* src, long long n) {
  long long i = (long long)blockIdx.x * 256 + threadIdx.x;
  if (i < n) dst[i] = (_Float16)src[i];
}

__global__ void k_sum3h(_Float16* dst, const float* a, const float* b, const float* c, long long n) {
  long long i = (long long)blockIdx.x * 256 + threadIdx.x;
  if (i < n) dst[i] = (_Float16)(a[i] + b[i] + c[i]);
}

__global__ void k_elu_h(_Float16* dst, const float* src, long long n) {
  long long i = (long long)blockIdx.x * 256 + threadIdx.x;
  if (i < n) { float v = src[i]; dst[i] = (_Float16)(v > 0.f ? v : expm1f(v)); }
}

// dst[bt][c][r] = src[bt][r][c]  (f32 -> f16)
__global__ void k_transpose_bt(_Float16* dst, const float* src, int batch, int rows, int cols) {
  long long idx = (long long)blockIdx.x * 256 + threadIdx.x;
  long long tot = (long long)batch * rows * cols;
  if (idx >= tot) return;
  int c = idx % cols;
  int r = (idx / cols) % rows;
  int bt = idx / ((long long)rows * cols);
  dst[((size_t)bt * cols + c) * rows + r] = (_Float16)src[idx];
}

__global__ void k_bnstats(const float* unb, const float* hpv,
                          const float* g1, const float* b1,
                          const float* gp, const float* bp, float* bnst) {
  int which = blockIdx.x >> 6, c = blockIdx.x & 63;
  const float* src = which ? hpv : unb;
  const float* g = which ? gp : g1;
  const float* bb = which ? bp : b1;
  __shared__ float s1[256], s2[256];
  int tid = threadIdx.x;
  float a = 0.f, b2 = 0.f;
  for (int i = tid; i < NBATCH*NNODE; i += 256) {
    float v = src[(size_t)i * 64 + c];
    a += v; b2 += v * v;
  }
  s1[tid] = a; s2[tid] = b2;
  __syncthreads();
  for (int o = 128; o > 0; o >>= 1) {
    if (tid < o) { s1[tid] += s1[tid+o]; s2[tid] += s2[tid+o]; }
    __syncthreads();
  }
  if (tid == 0) {
    float inv = 1.f / (NBATCH*NNODE);
    float mu = s1[0] * inv;
    float var = s2[0] * inv - mu * mu;
    float s = g[c] * rsqrtf(var + 1e-5f);
    bnst[which*128 + c] = s;
    bnst[which*128 + 64 + c] = bb[c] - mu * s;
  }
}

__global__ void k_bnapply(float* unb, float* hpv, _Float16* unb16, _Float16* hpv16,
                          const float* bnst) {
  long long idx = (long long)blockIdx.x * 256 + threadIdx.x;
  long long half = (long long)NBATCH*NNODE*64;
  if (idx >= 2*half) return;
  int which = idx >= half;
  long long j = idx - (which ? half : 0);
  int c = j & 63;
  float* src = which ? hpv : unb;
  _Float16* d16 = which ? hpv16 : unb16;
  float v = src[j] * bnst[which*128 + c] + bnst[which*128 + 64 + c];
  src[j] = v;
  d16[j] = (_Float16)v;
}

__global__ void k_softmax(_Float16* dst, const float* src, int cols, int dorelu) {
  size_t base = (size_t)blockIdx.x * cols;
  int tid = threadIdx.x;
  __shared__ float red[256];
  float m = -1e30f;
  for (int i = tid; i < cols; i += 256) {
    float v = src[base+i]; if (dorelu && v < 0.f) v = 0.f;
    if (v > m) m = v;
  }
  red[tid] = m; __syncthreads();
  for (int o = 128; o > 0; o >>= 1) {
    if (tid < o && red[tid+o] > red[tid]) red[tid] = red[tid+o];
    __syncthreads();
  }
  m = red[0]; __syncthreads();
  float s = 0.f;
  for (int i = tid; i < cols; i += 256) {
    float v = src[base+i]; if (dorelu && v < 0.f) v = 0.f;
    s += expf(v - m);
  }
  red[tid] = s; __syncthreads();
  for (int o = 128; o > 0; o >>= 1) {
    if (tid < o) red[tid] += red[tid+o];
    __syncthreads();
  }
  float inv = 1.f / red[0];
  for (int i = tid; i < cols; i += 256) {
    float v = src[base+i]; if (dorelu && v < 0.f) v = 0.f;
    dst[base+i] = (_Float16)(expf(v - m) * inv);
  }
}

__global__ void k_gather_te(const int* ind, const float* nvp1, _Float16* te16) {
  int idx = blockIdx.x * 256 + threadIdx.x;
  if (idx >= NBATCH*64) return;
  int b = idx >> 6, c = idx & 63;
  int i0 = ind[(b*12 + 1)*2 + 0];
  int i1 = ind[(b*12 + 1)*2 + 1];
  int r = (i0 <= 4) ? i1 : 23 + i1;
  te16[idx] = (_Float16)nvp1[r*64 + c];
}

// ---------------- host ----------------
static void launch_gemm(hipStream_t st, const _Float16* A, long long sA, int lda,
                        const _Float16* B, long long sB, int ldb,
                        float* C, long long sC, int ldc,
                        const float* bias, long long sBias,
                        int M, int N, int K, int batch, int beta) {
  dim3 g((M + 15) / 16, (N + 15) / 16, batch);
  k_gemm<<<g, 32, 0, st>>>(A, sA, lda, B, sB, ldb, C, sC, ldc, bias, sBias, M, N, K, beta);
}

extern "C" void kernel_launch(void* const* d_in, const int* in_sizes, int n_in,
                              void* d_out, int out_size, void* d_ws, size_t ws_size,
                              hipStream_t stream) {
  (void)in_sizes; (void)n_in; (void)out_size; (void)ws_size;
  const float* x     = (const float*)d_in[0];
  const float* adj   = (const float*)d_in[2];
  const float* fcxW  = (const float*)d_in[3];
  const float* fcxb  = (const float*)d_in[4];
  const float* env   = (const float*)d_in[5];
  const float* ncb   = (const float*)d_in[6];
  const float* wv    = (const float*)d_in[7];
  const float* wk    = (const float*)d_in[8];
  const float* wq    = (const float*)d_in[9];
  const float* coreW = (const float*)d_in[10];
  const float* coreb = (const float*)d_in[11];
  const float* WinN  = (const float*)d_in[12];
  const float* binN  = (const float*)d_in[13];
  const float* WfN   = (const float*)d_in[14];
  const float* bfN   = (const float*)d_in[15];
  const float* WgN   = (const float*)d_in[16];
  const float* bgN   = (const float*)d_in[17];
  const float* WoN   = (const float*)d_in[18];
  const float* boN   = (const float*)d_in[19];
  const float* WinP  = (const float*)d_in[20];
  const float* binP  = (const float*)d_in[21];
  const float* WfP   = (const float*)d_in[22];
  const float* bfP   = (const float*)d_in[23];
  const float* WgP   = (const float*)d_in[24];
  const float* bgP   = (const float*)d_in[25];
  const float* WoP   = (const float*)d_in[26];
  const float* boP   = (const float*)d_in[27];
  const float* bn1g  = (const float*)d_in[28];
  const float* bn1b  = (const float*)d_in[29];
  const float* bnPg  = (const float*)d_in[30];
  const float* bnPb  = (const float*)d_in[31];
  const float* g1W   = (const float*)d_in[32];
  const float* g1b   = (const float*)d_in[33];
  const float* g2W   = (const float*)d_in[34];
  const float* g2b   = (const float*)d_in[35];
  const float* gpW   = (const float*)d_in[36];
  const float* gpb   = (const float*)d_in[37];
  const float* nvp1  = (const float*)d_in[38];
  const float* nvp2  = (const float*)d_in[39];
  const float* nvp3  = (const float*)d_in[40];
  const float* nvpk  = (const float*)d_in[41];
  const float* oW1   = (const float*)d_in[42];
  const float* ob1   = (const float*)d_in[43];
  const float* oW2   = (const float*)d_in[44];
  const float* ob2   = (const float*)d_in[45];
  const int* tpos    = (const int*)d_in[46];
  const int* indarr  = (const int*)d_in[47];
  float* out = (float*)d_out;

  char* p = (char*)d_ws;
  auto alloc = [&](size_t bytes) -> void* {
    void* r = (void*)p;
    p += (bytes + 255) & ~(size_t)255;
    return r;
  };
  const long long BN64 = (long long)NBATCH*NNODE*64;   // 512000
  float*     matk   = (float*)alloc(640*4);
  float*     matv   = (float*)alloc(100*4);
  int*       sel    = (int*)alloc(NNODE*4);
  int*       texp   = (int*)alloc(MAXTILES*4);
  int*       tnod   = (int*)alloc(MAXTILES*16*4);
  int*       ntil   = (int*)alloc(64*4);
  float*     temb   = (float*)alloc(NBATCH*12*64*4);
  _Float16*  temb11 = (_Float16*)alloc(NBATCH*64*2);
  _Float16*  X5     = (_Float16*)alloc((size_t)NBATCH*NNODE*5*128*2);
  _Float16*  PW     = (_Float16*)alloc((size_t)200*SLOTH*2);
  float*     unb    = (float*)alloc(BN64*4);
  float*     hpv    = (float*)alloc(BN64*4);
  float*     bnst   = (float*)alloc(256*4);
  _Float16*  unb16  = (_Float16*)alloc(BN64*2);
  _Float16*  hpv16  = (_Float16*)alloc(BN64*2);
  _Float16*  hpvT16 = (_Float16*)alloc(BN64*2);
  _Float16*  adjh   = (_Float16*)alloc((size_t)NNODE*NNODE*2);
  _Float16*  coreW16= (_Float16*)alloc((size_t)64*4096*2);
  float*     coref  = (float*)alloc((size_t)NBATCH*4096*4);
  _Float16*  core16 = (_Float16*)alloc((size_t)NBATCH*4096*2);
  float*     t1f    = (float*)alloc(BN64*4);
  _Float16*  t1h    = (_Float16*)alloc(BN64*2);
  float*     ztb1   = (float*)alloc(BN64*4);
  float*     ztb2   = (float*)alloc(BN64*4);
  _Float16*  zth1   = (_Float16*)alloc(BN64*2);
  _Float16*  acc16  = (_Float16*)alloc(BN64*2);
  float*     nxn    = (float*)alloc((size_t)NBATCH*NNODE*NNODE*4);
  _Float16*  c16    = (_Float16*)alloc((size_t)NBATCH*NNODE*NNODE*2);
  _Float16*  te16   = (_Float16*)alloc(NBATCH*64*2);
  _Float16*  nvpk16 = (_Float16*)alloc((size_t)64*4096*2);
  float*     adp1f  = (float*)alloc((size_t)NBATCH*4096*4);
  _Float16*  adp1h  = (_Float16*)alloc((size_t)NBATCH*4096*2);
  _Float16*  nvp2h  = (_Float16*)alloc((size_t)NNODE*64*2);
  _Float16*  nvp3T  = (_Float16*)alloc((size_t)64*NNODE*2);
  _Float16*  g1w16  = (_Float16*)alloc(4096*2);
  _Float16*  g2w16  = (_Float16*)alloc(4096*2);
  _Float16*  gpw16  = (_Float16*)alloc(4096*2);
  float*     curh   = (float*)alloc(BN64*4);
  _Float16*  cur16  = (_Float16*)alloc(BN64*2);
  _Float16*  oW1h   = (_Float16*)alloc((size_t)12*64*100*2);
  _Float16*  oW2h   = (_Float16*)alloc((size_t)12*100*2*2);
  float*     hmidf  = (float*)alloc((size_t)NBATCH*NNODE*100*4);
  _Float16*  hmidh  = (_Float16*)alloc((size_t)NBATCH*NNODE*100*2);

  // ---- expert selection + tiling ----
  k_small0<<<1, 768, 0, stream>>>(env, wk, wv, matk, matv);
  k_sel<<<2, 256, 0, stream>>>(ncb, wq, matk, matv, sel);
  k_tiles<<<1, 512, 0, stream>>>(sel, texp, tnod, ntil);

  // ---- embeddings / inputs ----
  k_temb<<<(NBATCH*12*64 + 255)/256, 256, 0, stream>>>(tpos, temb, temb11);
  {
    long long tot = (long long)NBATCH*NNODE*5*128;
    k_totalx<<<(unsigned)((tot + 255)/256), 256, 0, stream>>>(x, fcxW, fcxb, temb, X5);
  }
  k_pack<<<dim3(10, 2, 10), 256, 0, stream>>>(WinN, WfN, WgN, WoN, WinP, WfP, WgP, WoP, PW);

  // ---- TCN expert mixture (WMMA) ----
  k_tcn<<<dim3(MAXTILES, NBATCH), 128, 0, stream>>>(X5, PW, texp, tnod, ntil,
      binN, bfN, bgN, boN, binP, bfP, bgP, boP, unb, hpv);

  // ---- batch norms ----
  k_bnstats<<<128, 256, 0, stream>>>(unb, hpv, bn1g, bn1b, bnPg, bnPb, bnst);
  k_bnapply<<<(unsigned)((2*BN64 + 255)/256), 256, 0, stream>>>(unb, hpv, unb16, hpv16, bnst);
  k_transpose_bt<<<(unsigned)((BN64 + 255)/256), 256, 0, stream>>>(hpvT16, hpv, NBATCH, NNODE, 64);

  // ---- f16 weight conversions ----
  k_f32h<<<(NNODE*NNODE + 255)/256, 256, 0, stream>>>(adjh, adj, (long long)NNODE*NNODE);
  k_f32h<<<(64*4096 + 255)/256, 256, 0, stream>>>(coreW16, coreW, 64*4096);
  k_f32h<<<(64*4096 + 255)/256, 256, 0, stream>>>(nvpk16, nvpk, 64*4096);
  k_f32h<<<(NNODE*64 + 255)/256, 256, 0, stream>>>(nvp2h, nvp2, NNODE*64);
  k_transpose_bt<<<(NNODE*64 + 255)/256, 256, 0, stream>>>(nvp3T, nvp3, 1, NNODE, 64);
  k_f32h<<<16, 256, 0, stream>>>(g1w16, g1W, 4096);
  k_f32h<<<16, 256, 0, stream>>>(g2w16, g2W, 4096);
  k_f32h<<<16, 256, 0, stream>>>(gpw16, gpW, 4096);
  k_f32h<<<(12*64*100 + 255)/256, 256, 0, stream>>>(oW1h, oW1, 12*64*100);
  k_f32h<<<(12*100*2 + 255)/256, 256, 0, stream>>>(oW2h, oW2, 12*100*2);

  // ---- GCN1: h1 = (h_prev + A h_prev + A^2 h_prev) @ W1 + b1 -> cur_h ----
  launch_gemm(stream, adjh, 0, NNODE, hpv16, 32000, 64, ztb1, 32000, 64,
              nullptr, 0, NNODE, 64, NNODE, NBATCH, 0);
  k_f32h<<<(unsigned)((BN64+255)/256), 256, 0, stream>>>(zth1, ztb1, BN64);
  launch_gemm(stream, adjh, 0, NNODE, zth1, 32000, 64, ztb2, 32000, 64,
              nullptr, 0, NNODE, 64, NNODE, NBATCH, 0);
  k_sum3h<<<(unsigned)((BN64+255)/256), 256, 0, stream>>>(acc16, hpv, ztb1, ztb2, BN64);
  launch_gemm(stream, acc16, 0, 64, g1w16, 0, 64, curh, 0, 64,
              g1b, 0, NBATCH*NNODE, 64, 64, 1, 0);

  // ---- causal attention adjacency + GCN2 ----
  launch_gemm(stream, temb11, 0, 64, coreW16, 0, 4096, coref, 0, 4096,
              coreb, 0, NBATCH, 4096, 64, 1, 0);
  k_f32h<<<(NBATCH*4096 + 255)/256, 256, 0, stream>>>(core16, coref, NBATCH*4096);
  launch_gemm(stream, unb16, 32000, 64, core16, 4096, 64, t1f, 32000, 64,
              nullptr, 0, NNODE, 64, 64, NBATCH, 0);
  k_f32h<<<(unsigned)((BN64+255)/256), 256, 0, stream>>>(t1h, t1f, BN64);
  launch_gemm(stream, t1h, 32000, 64, hpvT16, 32000, NNODE, nxn, (long long)NNODE*NNODE, NNODE,
              nullptr, 0, NNODE, NNODE, 64, NBATCH, 0);
  k_softmax<<<NBATCH*NNODE, 256, 0, stream>>>(c16, nxn, NNODE, 0);
  launch_gemm(stream, c16, (long long)NNODE*NNODE, NNODE, unb16, 32000, 64, ztb1, 32000, 64,
              nullptr, 0, NNODE, 64, NNODE, NBATCH, 0);
  k_f32h<<<(unsigned)((BN64+255)/256), 256, 0, stream>>>(zth1, ztb1, BN64);
  launch_gemm(stream, c16, (long long)NNODE*NNODE, NNODE, zth1, 32000, 64, ztb2, 32000, 64,
              nullptr, 0, NNODE, 64, NNODE, NBATCH, 0);
  k_sum3h<<<(unsigned)((BN64+255)/256), 256, 0, stream>>>(acc16, unb, ztb1, ztb2, BN64);
  launch_gemm(stream, acc16, 0, 64, g2w16, 0, 64, curh, 0, 64,
              g2b, 0, NBATCH*NNODE, 64, 64, 1, 1);

  // ---- adaptive adjacency + GCN3 ----
  k_gather_te<<<4, 256, 0, stream>>>(indarr, nvp1, te16);
  launch_gemm(stream, te16, 0, 64, nvpk16, 0, 4096, adp1f, 0, 4096,
              nullptr, 0, NBATCH, 4096, 64, 1, 0);
  k_f32h<<<(NBATCH*4096 + 255)/256, 256, 0, stream>>>(adp1h, adp1f, NBATCH*4096);
  launch_gemm(stream, nvp2h, 0, 64, adp1h, 4096, 64, ztb1, 32000, 64,
              nullptr, 0, NNODE, 64, 64, NBATCH, 0);
  k_f32h<<<(unsigned)((BN64+255)/256), 256, 0, stream>>>(zth1, ztb1, BN64);
  launch_gemm(stream, zth1, 32000, 64, nvp3T, 0, NNODE, nxn, (long long)NNODE*NNODE, NNODE,
              nullptr, 0, NNODE, NNODE, 64, NBATCH, 0);
  k_softmax<<<NBATCH*NNODE, 256, 0, stream>>>(c16, nxn, NNODE, 1);
  launch_gemm(stream, c16, (long long)NNODE*NNODE, NNODE, unb16, 32000, 64, ztb1, 32000, 64,
              nullptr, 0, NNODE, 64, NNODE, NBATCH, 0);
  k_f32h<<<(unsigned)((BN64+255)/256), 256, 0, stream>>>(zth1, ztb1, BN64);
  launch_gemm(stream, c16, (long long)NNODE*NNODE, NNODE, zth1, 32000, 64, ztb2, 32000, 64,
              nullptr, 0, NNODE, 64, NNODE, NBATCH, 0);
  k_sum3h<<<(unsigned)((BN64+255)/256), 256, 0, stream>>>(acc16, unb, ztb1, ztb2, BN64);
  launch_gemm(stream, acc16, 0, 64, gpw16, 0, 64, curh, 0, 64,
              gpb, 0, NBATCH*NNODE, 64, 64, 1, 1);

  // ---- output MLP per timestep ----
  k_f32h<<<(unsigned)((BN64+255)/256), 256, 0, stream>>>(cur16, curh, BN64);
  const long long BN100 = (long long)NBATCH*NNODE*100;
  for (int t = 0; t < 12; ++t) {
    launch_gemm(stream, cur16, 0, 64, oW1h + (size_t)t*6400, 0, 100, hmidf, 0, 100,
                ob1 + t*100, 0, NBATCH*NNODE, 100, 64, 1, 0);
    k_elu_h<<<(unsigned)((BN100+255)/256), 256, 0, stream>>>(hmidh, hmidf, BN100);
    launch_gemm(stream, hmidh, 0, 100, oW2h + (size_t)t*200, 0, 2, out + t*2, 0, 24,
                ob2 + t*2, 0, NBATCH*NNODE, 2, 100, 1, 0);
  }
}
